// CorrelationLayer_25692494365440
// MI455X (gfx1250) — compile-verified, matrix operand order checked by reading, not run on hardware
//
#include <hip/hip_runtime.h>
#include <hip/hip_bf16.h>

typedef _Float16 v4h  __attribute__((ext_vector_type(4)));
typedef _Float16 v8h  __attribute__((ext_vector_type(8)));
typedef _Float16 v16h __attribute__((ext_vector_type(16)));
typedef float    v8f  __attribute__((ext_vector_type(8)));
// __builtin_amdgcn_cvt_pkrtz returns the __fp16-based vector type:
typedef __fp16   h2   __attribute__((ext_vector_type(2)));
typedef __fp16   h4   __attribute__((ext_vector_type(4)));

#define RS        4
#define DD        81          // (2*RS+1)^2
#define CH        64
#define HH        256
#define WW        256
#define PIX       64          // pixels (w positions) per workgroup
#define NTHREADS  128         // 4 waves
#define PSTR      72          // f16 elements per pixel row in LDS (64 + 8 pad)
#define X2PIX     80          // 64 + 2*4 halo + 8 over-read pad for key tile 2

// LDS layout (bytes); dump region = 128 dwords (absorbs +9*ii address drift)
#define X1_BYTES  (PIX * PSTR * 2)                                   // 9216
#define X2_BYTES  (X2PIX * PSTR * 2)                                 // 11520
#define CV_BYTES  (PIX * DD * 4)                                     // 20736
#define SM_BYTES  (X1_BYTES + X2_BYTES + CV_BYTES + 2*PIX*4 + 128*4) // 42496

__device__ __forceinline__ v16h frag16(const _Float16* plo, const _Float16* phi) {
    v8h lo = *(const v8h*)plo;
    v8h hi = *(const v8h*)phi;
    return __builtin_shufflevector(lo, hi, 0,1,2,3,4,5,6,7,8,9,10,11,12,13,14,15);
}

__device__ __forceinline__ void store_pk4(_Float16* dst, float f0, float f1,
                                          float f2, float f3) {
    h2 p0 = __builtin_amdgcn_cvt_pkrtz(f0, f1);
    h2 p1 = __builtin_amdgcn_cvt_pkrtz(f2, f3);
    h4 q  = __builtin_shufflevector(p0, p1, 0, 1, 2, 3);
    *(h4*)dst = q;          // 8-byte LDS store
}

__global__ __launch_bounds__(NTHREADS)
void corr_softmax_wmma(const float* __restrict__ x1,
                       const float* __restrict__ x2,
                       float* __restrict__ out)
{
    extern __shared__ char smem[];
    _Float16* x1s = (_Float16*)smem;                       // [PIX][PSTR]
    _Float16* x2s = (_Float16*)(smem + X1_BYTES);          // [X2PIX][PSTR]
    float*    cvs = (float*)(smem + X1_BYTES + X2_BYTES);  // [PIX][DD] + mxv/inv + dump

    const int t    = threadIdx.x;
    const int lane = t & 31;
    const int wave = t >> 5;
    const int b    = blockIdx.z;
    const int h    = blockIdx.y;
    const int w0   = blockIdx.x * PIX;
    const size_t HW = (size_t)HH * WW;

    // ---- stage x1 tile: 4 channels/thread, packed f16, ds_store_b64 ---------
    #pragma unroll 2
    for (int it = 0; it < (PIX * CH) / (4 * NTHREADS); ++it) {   // 8 iters
        int idx = t + it * NTHREADS;          // [0, 1024)
        int wl  = idx & 63;
        int c   = (idx >> 6) << 2;            // 0,4,...,60
        const float* src = x1 + (((size_t)b * CH + c) * HH + h) * WW + (w0 + wl);
        float f0 = src[0];
        float f1 = src[HW];
        float f2 = src[2 * HW];
        float f3 = src[3 * HW];
        store_pk4(&x1s[wl * PSTR + c], f0, f1, f2, f3);
    }
    __syncthreads();

    // ---- per-wave A fragments (loaded once, reused for all 9 i) -------------
    const int wv0 = wave * 16;         // wave's first pixel (local)
    const int nn  = lane & 15;         // M row (A) / N column (B)
    const int hi8 = (lane >> 4) * 8;   // K-half select per ISA 16-bit layout
    v16h A0, A1;
    {
        const _Float16* pa = x1s + (wv0 + nn) * PSTR;
        A0 = frag16(pa + hi8,      pa + 16 + hi8);   // K = 0..31
        A1 = frag16(pa + 32 + hi8, pa + 48 + hi8);   // K = 32..63
    }

    // ---- loop-invariant band-extraction constants per (lane, r) -------------
    // For each (lane, r) at most one of {tile1, tile2} yields a valid j.
    const int base    = (lane >> 4) * 8;               // C-matrix row base
    const int dumpIdx = PIX * DD + 2 * PIX + lane;     // per-lane trash slot
    int  kk0[8], adr0[8];
    bool sel1[8];
    #pragma unroll
    for (int r = 0; r < 8; ++r) {
        int  m     = base + r;
        int  j1    = nn - m;                 // n_tot - m, tile 1
        int  j2    = j1 + 16;                // n_tot - m, tile 2
        bool v1    = (unsigned)j1 <= 2u * RS;
        bool v2    = (unsigned)j2 <= 2u * RS;
        bool valid = v1 || v2;
        int  j     = v1 ? j1 : j2;
        int  k0    = 49 - j;                 // ((9*i_disp - j_disp) mod 81) base
        kk0[r]  = valid ? k0 : -1000;        // invalid: never triggers wrap
        adr0[r] = valid ? ((wv0 + m) * DD + k0) : dumpIdx;
        sel1[r] = v1;
    }

    // ---- loop over vertical displacement i ----------------------------------
    #pragma unroll 1
    for (int ii = 0; ii < 2 * RS + 1; ++ii) {
        const int  row  = h - (ii - RS);
        const int  rowc = min(max(row, 0), HH - 1);
        const bool rok  = (row >= 0) && (row < HH);

        // stage x2 row segment, branchless clamped loads, packed f16
        #pragma unroll 2
        for (int it = 0; it < (X2PIX * CH) / (4 * NTHREADS); ++it) {  // 10 iters
            int idx = t + it * NTHREADS;        // [0, 1280)
            int c4  = idx / X2PIX;
            int wl  = idx - c4 * X2PIX;
            int c   = c4 << 2;
            int w   = w0 - RS + wl;
            int wc  = min(max(w, 0), WW - 1);
            bool ok = rok && ((unsigned)w < (unsigned)WW);
            const float* src = x2 + (((size_t)b * CH + c) * HH + rowc) * WW + wc;
            float f0 = src[0];
            float f1 = src[HW];
            float f2 = src[2 * HW];
            float f3 = src[3 * HW];
            f0 = ok ? f0 : 0.0f;
            f1 = ok ? f1 : 0.0f;
            f2 = ok ? f2 : 0.0f;
            f3 = ok ? f3 : 0.0f;
            store_pk4(&x2s[wl * PSTR + c], f0, f1, f2, f3);
        }
        __syncthreads();

        // B fragments: key tile 1 (n_tot = nn) and key tile 2 (n_tot = nn+16)
        const _Float16* pb1 = x2s + (wv0 + nn) * PSTR;
        const _Float16* pb2 = x2s + (wv0 + 16 + nn) * PSTR;
        v16h B0a = frag16(pb1 + hi8,      pb1 + 16 + hi8);
        v16h B1a = frag16(pb1 + 32 + hi8, pb1 + 48 + hi8);
        v16h B0b = frag16(pb2 + hi8,      pb2 + 16 + hi8);
        v16h B1b = frag16(pb2 + 32 + hi8, pb2 + 48 + hi8);

        v8f acc1 = {0.f,0.f,0.f,0.f,0.f,0.f,0.f,0.f};
        v8f acc2 = {0.f,0.f,0.f,0.f,0.f,0.f,0.f,0.f};
        acc1 = __builtin_amdgcn_wmma_f32_16x16x32_f16(false, A0, false, B0a,
                                                      (short)0, acc1, false, false);
        acc1 = __builtin_amdgcn_wmma_f32_16x16x32_f16(false, A1, false, B1a,
                                                      (short)0, acc1, false, false);
        acc2 = __builtin_amdgcn_wmma_f32_16x16x32_f16(false, A0, false, B0b,
                                                      (short)0, acc2, false, false);
        acc2 = __builtin_amdgcn_wmma_f32_16x16x32_f16(false, A1, false, B1b,
                                                      (short)0, acc2, false, false);

        // ---- branchless band extraction: 8 unconditional ds_store_b32 -------
        const int s9 = 9 * ii;
        #pragma unroll
        for (int r = 0; r < 8; ++r) {
            int   kk  = kk0[r] + s9;
            int   adr = adr0[r] + s9;
            adr = (kk >= DD) ? adr - DD : adr;     // mod-81 wrap
            float val = sel1[r] ? acc1[r] : acc2[r];
            cvs[adr] = val;
        }
        __syncthreads();   // protect x2s for next iteration / cvs for softmax
    }

    // ---- softmax over the 81 channels ---------------------------------------
    float* mxv = cvs + PIX * DD;
    float* inv = mxv + PIX;
    if (t < PIX) {
        const float* p = cvs + t * DD;
        float m = -3.0e38f;
        #pragma unroll 1
        for (int k = 0; k < DD; ++k) m = fmaxf(m, p[k]);
        float s = 0.0f;
        #pragma unroll 1
        for (int k = 0; k < DD; ++k) s += __expf(p[k] - m);
        mxv[t] = m;
        inv[t] = 1.0f / s;
    }
    __syncthreads();

    #pragma unroll 1
    for (int idx = t; idx < DD * PIX; idx += NTHREADS) {
        int k  = idx >> 6;            // / PIX
        int wl = idx & 63;            // % PIX
        float v = __expf(cvs[wl * DD + k] - mxv[wl]) * inv[wl];
        __builtin_nontemporal_store(
            v, out + (((size_t)b * DD + k) * HH + h) * WW + (w0 + wl));
    }
}

extern "C" void kernel_launch(void* const* d_in, const int* in_sizes, int n_in,
                              void* d_out, int out_size, void* d_ws, size_t ws_size,
                              hipStream_t stream) {
    const float* x1 = (const float*)d_in[0];
    const float* x2 = (const float*)d_in[1];
    float* out = (float*)d_out;
    (void)d_ws; (void)ws_size; (void)out_size; (void)n_in;

    const int B = in_sizes[0] / (CH * HH * WW);   // = 4
    dim3 grid(WW / PIX, HH, B);                   // (4, 256, 4)
    dim3 block(NTHREADS);
    corr_softmax_wmma<<<grid, block, SM_BYTES, stream>>>(x1, x2, out);
}
